// MultiHeadAttention_60833916780574
// MI455X (gfx1250) — compile-verified, matrix operand order checked by reading, not run on hardware
//
#include <hip/hip_runtime.h>
#include <hip/hip_bf16.h>

// MHA: B=4, S=2048, D=1024, H=16, DK=64
#define BB 4
#define SS 2048
#define DD 1024
#define HH 16
#define DKK 64

typedef __attribute__((ext_vector_type(16))) _Float16 v16h;
typedef __attribute__((ext_vector_type(8)))  _Float16 v8h;
typedef __attribute__((ext_vector_type(8)))  float    v8f;
typedef __attribute__((ext_vector_type(4)))  int      v4i;

#define AS1 __attribute__((address_space(1)))
#define AS3 __attribute__((address_space(3)))

#if defined(__has_builtin)
#if __has_builtin(__builtin_amdgcn_global_load_async_to_lds_b128)
#define ASYNC_LDS 1
#endif
#endif
#ifndef ASYNC_LDS
#define ASYNC_LDS 0
#endif

// 16-byte global -> LDS copy. Async (ASYNCcnt-tracked, no VGPR data) when the
// gfx1250 builtin is available, otherwise synchronous through VGPRs.
// Builtin signature (from clang diagnostic): arg0 = v4i* in global AS,
// 4 args total (src, lds_dst, imm offset, imm cpol).
__device__ __forceinline__ void async_cp16(void* lds, const void* g) {
#if ASYNC_LDS
    __builtin_amdgcn_global_load_async_to_lds_b128(
        (AS1 v4i*)(unsigned long long)(size_t)g,
        (AS3 v4i*)(unsigned int)(unsigned long long)(size_t)lds,
        0, 0);
#else
    *(v8h*)lds = *(const v8h*)g;
#endif
}

__device__ __forceinline__ void wait_async() {
#if ASYNC_LDS
#if __has_builtin(__builtin_amdgcn_s_wait_asynccnt)
    __builtin_amdgcn_s_wait_asynccnt(0);
#else
    asm volatile("s_wait_asynccnt 0x0" ::: "memory");
#endif
#endif
}

// Fragment loader for 16-bit A (and symmetric B) WMMA operands, contiguous k order:
// lanes 0-15 hold k = kbase..kbase+15 for row/col (lane&15),
// lanes 16-31 hold k = kbase+16..kbase+31.
__device__ __forceinline__ v16h frag_ld(const _Float16* tile, int stride, int lane, int kbase) {
    const _Float16* p = tile + (lane & 15) * stride + kbase + ((lane >> 4) << 4);
    return *(const v16h*)p;
}

// ---------------------------------------------------------------------------
// Elementwise f32 -> f16 (vectorized x8). n must be a multiple of 2048.
// ---------------------------------------------------------------------------
__global__ __launch_bounds__(256) void cvt_f32_to_f16(
    const float* __restrict__ src, _Float16* __restrict__ dst)
{
    const size_t i = ((size_t)blockIdx.x * 256 + threadIdx.x) * 8;
    float4 a = *(const float4*)(src + i);
    float4 b = *(const float4*)(src + i + 4);
    v8h h = { (_Float16)a.x, (_Float16)a.y, (_Float16)a.z, (_Float16)a.w,
              (_Float16)b.x, (_Float16)b.y, (_Float16)b.z, (_Float16)b.w };
    *(v8h*)(dst + i) = h;
}

// ---------------------------------------------------------------------------
// GEMM:  out = (A @ Bw^T + bias) * scale      (A: [M,K] f16, Bw: [N,K] f16)
// MODE 0: f16 out [B,H,S,DK]  | MODE 1: f16 out [B,H,DK,S] | MODE 2: f32 [M,N]
// 256 threads (8 waves), 128x128 tile, K staged 64 at a time, double-buffered
// LDS filled with async global->LDS b128 copies. Wave grid 4(M) x 2(N); each
// wave computes 2x4 WMMA tiles => 16 v_wmma per stage per wave.
// ---------------------------------------------------------------------------
template <int MODE>
__global__ __launch_bounds__(256) void mha_gemm_wmma(
    const _Float16* __restrict__ A, const _Float16* __restrict__ Bw,
    const float* __restrict__ bias, void* __restrict__ outp,
    int M, int N, int K, float scale)
{
    __shared__ _Float16 ldsA[2][128 * 64];
    __shared__ _Float16 ldsB[2][128 * 64];

    const int tid  = threadIdx.x;
    const int lane = tid & 31;
    const int wid  = tid >> 5;
    const int wm   = wid & 3;
    const int wn   = wid >> 2;
    const int m0   = blockIdx.y * 128;
    const int n0   = blockIdx.x * 128;

    // Per-thread staging slice: 64 bytes of A tile + 64 bytes of B tile.
    const int srow = tid >> 1;          // 0..127
    const int sseg = (tid & 1) * 32;    // halves
    const _Float16* ga = A  + (size_t)(m0 + srow) * K + sseg;
    const _Float16* gb = Bw + (size_t)(n0 + srow) * K + sseg;

    auto issue_stage = [&](int ks, int buf) {
        const int koff = ks * 64;
#pragma unroll
        for (int j = 0; j < 4; j++) {
            async_cp16(&ldsA[buf][srow * 64 + sseg + j * 8], ga + koff + j * 8);
            async_cp16(&ldsB[buf][srow * 64 + sseg + j * 8], gb + koff + j * 8);
        }
    };

    v8f acc[2][4] = {};

    const int nst = K / 64;
    issue_stage(0, 0);
    wait_async();
    __syncthreads();

    for (int ks = 0; ks < nst; ks++) {
        const int cur = ks & 1;
        if (ks + 1 < nst) issue_stage(ks + 1, 1 - cur);   // prefetch next stage

#pragma unroll
        for (int c = 0; c < 2; c++) {                     // two k-chunks of 32
            v16h a[2], b[4];
#pragma unroll
            for (int mi = 0; mi < 2; mi++)
                a[mi] = frag_ld(&ldsA[cur][(wm * 32 + mi * 16) * 64], 64, lane, c * 32);
#pragma unroll
            for (int ni = 0; ni < 4; ni++)
                b[ni] = frag_ld(&ldsB[cur][(wn * 64 + ni * 16) * 64], 64, lane, c * 32);
#pragma unroll
            for (int mi = 0; mi < 2; mi++)
#pragma unroll
                for (int ni = 0; ni < 4; ni++)
                    acc[mi][ni] = __builtin_amdgcn_wmma_f32_16x16x32_f16(
                        false, a[mi], false, b[ni], (short)0, acc[mi][ni], false, false);
        }

        if (ks + 1 < nst) wait_async();                   // next stage landed
        __syncthreads();                                  // everyone done reading cur
    }

    // Epilogue. C layout: VGPR i -> rows i (lanes 0-15) / i+8 (lanes 16-31).
    const int hi = lane >> 4;
    const int ln = lane & 15;
#pragma unroll
    for (int mi = 0; mi < 2; mi++) {
#pragma unroll
        for (int ni = 0; ni < 4; ni++) {
            const int col = n0 + wn * 64 + ni * 16 + ln;
            const float bv = bias[col];
#pragma unroll
            for (int i = 0; i < 8; i++) {
                const int row = m0 + wm * 32 + mi * 16 + i + hi * 8;
                const float val = (acc[mi][ni][i] + bv) * scale;
                if constexpr (MODE == 2) {
                    ((float*)outp)[(size_t)row * N + col] = val;
                } else {
                    const int b  = row / SS;
                    const int s  = row - b * SS;
                    const int h  = col >> 6;     // DK = 64
                    const int dk = col & 63;
                    size_t idx;
                    if constexpr (MODE == 0)
                        idx = ((size_t)((b * HH + h) * SS + s)) * DKK + dk;
                    else
                        idx = ((size_t)((b * HH + h) * DKK + dk)) * SS + s;
                    ((_Float16*)outp)[idx] = (_Float16)val;
                }
            }
        }
    }
}

// ---------------------------------------------------------------------------
// Flash attention core.  Qp,Kp: [B,H,S,DK] f16 (Q pre-scaled by 1/sqrt(DK)),
// Vt: [B,H,DK,S] f16.  Output: f16 [B,S,D] (heads re-merged) for the final
// f16 projection GEMM.  128 threads (4 waves), one block per
// (b,h, 64-row query tile); causal key loop stops at the diagonal.
// ---------------------------------------------------------------------------
__global__ __launch_bounds__(128) void mha_flash_attn_wmma(
    const _Float16* __restrict__ Qp, const _Float16* __restrict__ Kp,
    const _Float16* __restrict__ Vt, _Float16* __restrict__ attn)
{
    __shared__ _Float16 ldsK[64 * 64];     // key tile  [key][dk]
    __shared__ _Float16 ldsV[64 * 64];     // V^T tile  [dk][key]
    __shared__ _Float16 ldsP[4][16 * 64];  // per-wave P tile [qrow][key]

    const int tid  = threadIdx.x;
    const int lane = tid & 31;
    const int wid  = tid >> 5;       // 0..3
    const int hi   = lane >> 4;
    const int ln   = lane & 15;

    const int qtile = blockIdx.x;
    const int bh    = blockIdx.y;
    const int b     = bh / HH;
    const int h     = bh - b * HH;
    const int qrow0 = qtile * 64 + wid * 16;

    const _Float16* Qbh = Qp + (size_t)bh * SS * DKK;
    const _Float16* Kbh = Kp + (size_t)bh * SS * DKK;
    const _Float16* Vbh = Vt + (size_t)bh * DKK * SS;

    // This wave's Q fragments: 16 rows x DK=64 (two k-chunks of 32).
    v16h qf[2];
#pragma unroll
    for (int c = 0; c < 2; c++)
        qf[c] = *(const v16h*)(Qbh + (size_t)(qrow0 + ln) * DKK + c * 32 + hi * 16);

    v8f o[4] = {};
    float rmax[8], rsum[8];
#pragma unroll
    for (int i = 0; i < 8; i++) { rmax[i] = -3.0e38f; rsum[i] = 0.0f; }

    for (int kt = 0; kt <= qtile; kt++) {
        const int kt0 = kt * 64;
        __syncthreads();
        // Async-stage K tile (contiguous rows) and V^T tile (row stride S).
        {
            const _Float16* ksrc = Kbh + (size_t)kt0 * DKK + tid * 32;
#pragma unroll
            for (int j = 0; j < 4; j++)
                async_cp16(&ldsK[tid * 32 + j * 8], ksrc + j * 8);
            const int d  = tid >> 1;
            const int cs = (tid & 1) * 32;
            const _Float16* vsrc = Vbh + (size_t)d * SS + kt0 + cs;
#pragma unroll
            for (int j = 0; j < 4; j++)
                async_cp16(&ldsV[d * 64 + cs + j * 8], vsrc + j * 8);
        }
        wait_async();
        __syncthreads();

        // Scores: 16 x 64 = 4 WMMA tiles, K-dim = DK = 64 (2 chunks).
        v8f sc[4] = {};
#pragma unroll
        for (int t = 0; t < 4; t++) {
#pragma unroll
            for (int c = 0; c < 2; c++) {
                v16h kf = frag_ld(ldsK + (t * 16) * 64, 64, lane, c * 32);
                sc[t] = __builtin_amdgcn_wmma_f32_16x16x32_f16(
                    false, qf[c], false, kf, (short)0, sc[t], false, false);
            }
        }

        // Causal mask + online softmax (each row's 64 scores live in one
        // 16-lane half => reduce with xor shuffles 1/2/4/8).
#pragma unroll
        for (int i = 0; i < 8; i++) {
            const int row = qrow0 + i + hi * 8;
            float mnew = rmax[i];
#pragma unroll
            for (int t = 0; t < 4; t++) {
                const int col = kt0 + t * 16 + ln;
                float s = (col <= row) ? sc[t][i] : -1.0e9f;
                sc[t][i] = s;
                mnew = fmaxf(mnew, s);
            }
#pragma unroll
            for (int msk = 1; msk < 16; msk <<= 1)
                mnew = fmaxf(mnew, __shfl_xor(mnew, msk, 32));
            const float corr = __expf(rmax[i] - mnew);
            rmax[i] = mnew;
            float ssum = 0.0f;
#pragma unroll
            for (int t = 0; t < 4; t++) {
                const float p = __expf(sc[t][i] - mnew);
                sc[t][i] = p;
                ssum += p;
            }
#pragma unroll
            for (int msk = 1; msk < 16; msk <<= 1)
                ssum += __shfl_xor(ssum, msk, 32);
            rsum[i] = rsum[i] * corr + ssum;
#pragma unroll
            for (int t = 0; t < 4; t++) o[t][i] *= corr;
        }

        // P (f32, C-layout) -> per-wave LDS as f16 A-tile [16 x 64].
#pragma unroll
        for (int t = 0; t < 4; t++)
#pragma unroll
            for (int i = 0; i < 8; i++)
                ldsP[wid][(i + hi * 8) * 64 + t * 16 + ln] = (_Float16)sc[t][i];
        __syncthreads();

        // O += P @ V^T  (4 WMMA tiles over DK, 2 key-chunks of 32).
#pragma unroll
        for (int c = 0; c < 2; c++) {
            v16h pf = frag_ld(&ldsP[wid][0], 64, lane, c * 32);
#pragma unroll
            for (int t = 0; t < 4; t++) {
                v16h vf = frag_ld(ldsV + (t * 16) * 64, 64, lane, c * 32);
                o[t] = __builtin_amdgcn_wmma_f32_16x16x32_f16(
                    false, pf, false, vf, (short)0, o[t], false, false);
            }
        }
    }

    // Normalize and store merged-head f16 output [B,S,D].
#pragma unroll
    for (int i = 0; i < 8; i++) {
        const int row = qrow0 + i + hi * 8;
        const float inv = 1.0f / rsum[i];
#pragma unroll
        for (int t = 0; t < 4; t++) {
            const int col = h * 64 + t * 16 + ln;
            attn[((size_t)(b * SS) + row) * DD + col] = (_Float16)(o[t][i] * inv);
        }
    }
}

extern "C" void kernel_launch(void* const* d_in, const int* in_sizes, int n_in,
                              void* d_out, int out_size, void* d_ws, size_t ws_size,
                              hipStream_t stream) {
    (void)in_sizes; (void)n_in; (void)out_size; (void)ws_size;

    const float* Q  = (const float*)d_in[0];
    const float* K  = (const float*)d_in[1];
    const float* V  = (const float*)d_in[2];
    // d_in[3] = causal mask (applied analytically in-kernel)
    const float* Wq = (const float*)d_in[4];
    const float* bq = (const float*)d_in[5];
    const float* Wk = (const float*)d_in[6];
    const float* bk = (const float*)d_in[7];
    const float* Wv = (const float*)d_in[8];
    const float* bv = (const float*)d_in[9];
    const float* Wo = (const float*)d_in[10];
    const float* bo = (const float*)d_in[11];

    // Workspace (all f16): Qh|Kh|Vh (16MB) Wqh|Wkh|Wvh|Woh (2MB)
    //                      Qp|Kp|Vt (16MB) attnh (16MB)  => 120MB
    char* ws = (char*)d_ws;
    const size_t ne = (size_t)BB * SS * DD;   // 8,388,608
    const size_t nw = (size_t)DD * DD;        // 1,048,576
    _Float16* Qh  = (_Float16*)ws; ws += ne * 2;
    _Float16* Kh  = (_Float16*)ws; ws += ne * 2;
    _Float16* Vh  = (_Float16*)ws; ws += ne * 2;
    _Float16* Wqh = (_Float16*)ws; ws += nw * 2;
    _Float16* Wkh = (_Float16*)ws; ws += nw * 2;
    _Float16* Wvh = (_Float16*)ws; ws += nw * 2;
    _Float16* Woh = (_Float16*)ws; ws += nw * 2;
    _Float16* Qp  = (_Float16*)ws; ws += ne * 2;
    _Float16* Kp  = (_Float16*)ws; ws += ne * 2;
    _Float16* Vt  = (_Float16*)ws; ws += ne * 2;
    _Float16* ah  = (_Float16*)ws;

    // f32 -> f16 conversions (inputs + weights)
    cvt_f32_to_f16<<<dim3(ne / 2048), 256, 0, stream>>>(Q,  Qh);
    cvt_f32_to_f16<<<dim3(ne / 2048), 256, 0, stream>>>(K,  Kh);
    cvt_f32_to_f16<<<dim3(ne / 2048), 256, 0, stream>>>(V,  Vh);
    cvt_f32_to_f16<<<dim3(nw / 2048), 256, 0, stream>>>(Wq, Wqh);
    cvt_f32_to_f16<<<dim3(nw / 2048), 256, 0, stream>>>(Wk, Wkh);
    cvt_f32_to_f16<<<dim3(nw / 2048), 256, 0, stream>>>(Wv, Wvh);
    cvt_f32_to_f16<<<dim3(nw / 2048), 256, 0, stream>>>(Wo, Woh);

    const int M = BB * SS, N = DD, Kd = DD;
    dim3 gridG(N / 128, M / 128);   // (8, 64)
    dim3 gridA(SS / 64, BB * HH);   // (32, 64)

    // Projections (Q pre-scaled by 1/sqrt(DK) = 1/8; V stored transposed)
    mha_gemm_wmma<0><<<gridG, 256, 0, stream>>>(Qh, Wqh, bq, (void*)Qp, M, N, Kd, 0.125f);
    mha_gemm_wmma<0><<<gridG, 256, 0, stream>>>(Kh, Wkh, bk, (void*)Kp, M, N, Kd, 1.0f);
    mha_gemm_wmma<1><<<gridG, 256, 0, stream>>>(Vh, Wvh, bv, (void*)Vt, M, N, Kd, 1.0f);
    // Causal flash attention -> f16 merged heads
    mha_flash_attn_wmma<<<gridA, 128, 0, stream>>>(Qp, Kp, Vt, ah);
    // Output projection -> d_out (f32)
    mha_gemm_wmma<2><<<gridG, 256, 0, stream>>>(ah, Woh, bo, d_out, M, N, Kd, 1.0f);
}
